// TransformerModel_36876589203668
// MI455X (gfx1250) — compile-verified, matrix operand order checked by reading, not run on hardware
//
#include <hip/hip_runtime.h>
#include <hip/hip_bf16.h>

#define USE_ASYNC 1

typedef __attribute__((ext_vector_type(16))) _Float16 v16h;
typedef __attribute__((ext_vector_type(8)))  _Float16 v8h;
typedef __attribute__((ext_vector_type(8)))  float    v8f;

static constexpr int D_     = 512;
static constexpr int LAYERS = 12;
static constexpr int SEQ    = 2048;
static constexpr int TOK    = 8192;   // 4 * 2048
static constexpr int DH     = 64;
static constexpr int DFF_   = 2048;
static constexpr int EMBW   = 1376;

// ---------------------------------------------------------------------------
// WMMA fragment load: 16x32 f16 tile, row-major with given element stride.
// Lane l in [0,15]  holds row (l),    K = 0..7  and 16..23
// Lane l in [16,31] holds row (l-16), K = 8..15 and 24..31
// ---------------------------------------------------------------------------
__device__ inline v16h frag_ld(const _Float16* base, int stride) {
  const int lane = threadIdx.x & 31;
  const _Float16* row = base + (lane & 15) * stride + ((lane >> 4) << 3);
  v8h lo = *(const v8h*)(row);
  v8h hi = *(const v8h*)(row + 16);
  v16h r;
#pragma unroll
  for (int i = 0; i < 8; ++i) { r[i] = lo[i]; r[i + 8] = hi[i]; }
  return r;
}

#if USE_ASYNC
// Async global -> LDS copy, 16 bytes per lane, ASYNCcnt-tracked (no VGPR data).
__device__ inline void async_b128(const void* g, void* lds) {
  const unsigned int l = (unsigned int)(size_t)lds;   // low 32b of generic = LDS byte addr
  asm volatile("global_load_async_to_lds_b128 %0, %1, off"
               :: "v"(l), "v"(g) : "memory");
}
__device__ inline void wait_async0() {
  asm volatile("s_wait_asynccnt 0x0" ::: "memory");
}
#endif

enum { EPI_F32 = 0, EPI_POSENC = 1, EPI_ELU1 = 2, EPI_F16 = 3,
       EPI_GELU = 4, EPI_RESCLIP = 5 };

// ---------------------------------------------------------------------------
// Tiled WMMA GEMM:  C[M,N] = A[M,K] (f16, row-major) x Bt[N,K]^T (f16)
// Block tile 128 x TBN (TBN = 128 main path, 64 for the small attn-num GEMM),
// K-tile 32, 256 threads = 8 waves arranged 4(M) x 2(N); wave tile 32 x TBN/2.
// Double-buffered LDS filled by async global->LDS loads.
// ---------------------------------------------------------------------------
template<int EPI, int TBN>
__global__ __launch_bounds__(256)
void k_gemm(const _Float16* __restrict__ A, int lda,
            const _Float16* __restrict__ Bt, int ldb,
            const float* __restrict__ bias,
            const float* __restrict__ resid,
            float* __restrict__ Cf, _Float16* __restrict__ Ch,
            int ldc, int K)
{
  constexpr int NF = TBN / 32;                 // B fragments per wave (2 or 4)
  __shared__ __align__(16) _Float16 sA[2][128][40];   // 80B rows, 16B aligned
  __shared__ __align__(16) _Float16 sB[2][TBN][40];

  const int tid  = threadIdx.x;
  const int lane = tid & 31, wid = tid >> 5;
  const int bM = blockIdx.y * 128, bN = blockIdx.x * TBN;
  const int wM = (wid >> 1) * 32,  wN = (wid & 1) * (TBN / 2);

  v8f acc[2][NF] = {};

  // staging maps: A tile 128x32 -> 16 f16/thread; B tile TBNx32
  const int ar = tid >> 1, ac = (tid & 1) * 16;
  const int br = (TBN == 128) ? (tid >> 1) : (tid >> 2);
  const int bc = (TBN == 128) ? (tid & 1) * 16 : (tid & 3) * 8;
  const _Float16* ag = A  + (size_t)(bM + ar) * lda + ac;
  const _Float16* bg = Bt + (size_t)(bN + br) * ldb + bc;

  const int nk = K / 32;

#if USE_ASYNC
  auto stage = [&](int buf, const _Float16* agp, const _Float16* bgp) {
    async_b128(agp,     &sA[buf][ar][ac]);
    async_b128(agp + 8, &sA[buf][ar][ac + 8]);
    async_b128(bgp, &sB[buf][br][bc]);
    if constexpr (TBN == 128) async_b128(bgp + 8, &sB[buf][br][bc + 8]);
  };
  stage(0, ag, bg);
#else
  auto stage = [&](int buf, const _Float16* agp, const _Float16* bgp) {
    *(v8h*)&sA[buf][ar][ac]     = *(const v8h*)(agp);
    *(v8h*)&sA[buf][ar][ac + 8] = *(const v8h*)(agp + 8);
    *(v8h*)&sB[buf][br][bc]     = *(const v8h*)(bgp);
    if constexpr (TBN == 128) *(v8h*)&sB[buf][br][bc + 8] = *(const v8h*)(bgp + 8);
  };
  stage(0, ag, bg);
#endif

  int buf = 0;
  for (int i = 0; i < nk; ++i) {
#if USE_ASYNC
    wait_async0();                     // this wave's fill of sX[buf] landed
#endif
    __syncthreads();                   // everyone's fill landed
    if (i + 1 < nk) {                  // fill the other buffer while computing
      ag += 32; bg += 32;
      stage(buf ^ 1, ag, bg);
    }

    v16h bf[NF];
#pragma unroll
    for (int ni = 0; ni < NF; ++ni)
      bf[ni] = frag_ld(&sB[buf][wN + ni * 16][0], 40);
#pragma unroll
    for (int mi = 0; mi < 2; ++mi) {
      v16h af = frag_ld(&sA[buf][wM + mi * 16][0], 40);
#pragma unroll
      for (int ni = 0; ni < NF; ++ni) {
        acc[mi][ni] = __builtin_amdgcn_wmma_f32_16x16x32_f16(
            false, af, false, bf[ni], (short)0, acc[mi][ni], false, false);
      }
    }
    __syncthreads();                   // reads of sX[buf] done before refill (i+2)
    buf ^= 1;
  }

  // C/D layout: VGPR r -> row r (lanes 0-15) or r+8 (lanes 16-31); col = lane&15
  const int rhi = (lane >> 4) * 8;
  const int cn  = lane & 15;
#pragma unroll
  for (int mi = 0; mi < 2; ++mi)
#pragma unroll
  for (int ni = 0; ni < NF; ++ni) {
#pragma unroll
    for (int r = 0; r < 8; ++r) {
      const int m = bM + wM + mi * 16 + rhi + r;
      const int n = bN + wN + ni * 16 + cn;
      float v = acc[mi][ni][r];
      if (bias) v += bias[n];
      const size_t o = (size_t)m * ldc + n;
      if constexpr (EPI == EPI_F32) {
        Cf[o] = v;
      } else if constexpr (EPI == EPI_POSENC) {
        const float pos = (float)(m & (SEQ - 1));
        const float ang = pos * __expf(-(float)(n & ~1) * (9.210340372f / 512.0f));
        v += (n & 1) ? __cosf(ang) : __sinf(ang);
        Cf[o] = v;
      } else if constexpr (EPI == EPI_ELU1) {          // elu(x)+1
        Ch[o] = (_Float16)((v > 0.0f) ? v + 1.0f : __expf(v));
      } else if constexpr (EPI == EPI_F16) {
        Ch[o] = (_Float16)v;
      } else if constexpr (EPI == EPI_GELU) {          // exact gelu
        Ch[o] = (_Float16)(0.5f * v * (1.0f + erff(v * 0.70710678f)));
      } else {                                         // residual + clip, f32
        float t = resid[o] + v;
        t = fminf(100.0f, fmaxf(-100.0f, t));
        Cf[o] = t;
      }
    }
  }
}

// f32 [K,N] -> f16 transposed [N,K] (WMMA B operand layout)
__global__ __launch_bounds__(256)
void k_cvtT(const float* __restrict__ src, _Float16* __restrict__ dst, int K, int N)
{
  const size_t idx = (size_t)blockIdx.x * 256 + threadIdx.x;
  if (idx >= (size_t)K * N) return;
  const int n = (int)(idx / K), k = (int)(idx % K);
  dst[idx] = (_Float16)src[(size_t)k * N + n];
}

struct EmbTabs { const float* t[8]; };

// gather 8 embedding fields, scale by sqrt(width), clip, concat -> f16 [TOK,1376]
__global__ __launch_bounds__(256)
void k_embed(const int* __restrict__ x, EmbTabs tabs, _Float16* __restrict__ E)
{
  const int tok = blockIdx.x;
  const int* xr = x + tok * 8;
  for (int c = threadIdx.x; c < EMBW; c += 256) {
    int f, off, wdt; float sc;
    if      (c < 128)  { f = 0; off = 0;    wdt = 128; sc = 11.3137085f; }
    else if (c < 384)  { f = 1; off = 128;  wdt = 256; sc = 16.0f;       }
    else if (c < 448)  { f = 2; off = 384;  wdt = 64;  sc = 8.0f;        }
    else if (c < 480)  { f = 3; off = 448;  wdt = 32;  sc = 5.65685425f; }
    else if (c < 992)  { f = 4; off = 480;  wdt = 512; sc = 22.627417f;  }
    else if (c < 1120) { f = 5; off = 992;  wdt = 128; sc = 11.3137085f; }
    else if (c < 1248) { f = 6; off = 1120; wdt = 128; sc = 11.3137085f; }
    else               { f = 7; off = 1248; wdt = 128; sc = 11.3137085f; }
    float v = tabs.t[f][(size_t)xr[f] * wdt + (c - off)] * sc;
    v = fminf(10.0f, fmaxf(-10.0f, v));
    E[(size_t)tok * EMBW + c] = (_Float16)v;
  }
}

// LayerNorm over D=512; writes f16 (and optionally f32 for the final LN)
template<bool WF32>
__global__ __launch_bounds__(256)
void k_ln(const float* __restrict__ in, const float* __restrict__ sc,
          const float* __restrict__ bi, _Float16* __restrict__ outH,
          float* __restrict__ outF)
{
  const int s = blockIdx.x;
  const int tid = threadIdx.x, lane = tid & 31, w = tid >> 5;
  const float* row = in + (size_t)s * D_;
  const float v0 = row[tid], v1 = row[tid + 256];
  float s1 = v0 + v1, s2 = v0 * v0 + v1 * v1;
#pragma unroll
  for (int off = 16; off; off >>= 1) {
    s1 += __shfl_xor(s1, off, 32);
    s2 += __shfl_xor(s2, off, 32);
  }
  __shared__ float a1[8], a2[8];
  if (lane == 0) { a1[w] = s1; a2[w] = s2; }
  __syncthreads();
  float t1 = 0.f, t2 = 0.f;
#pragma unroll
  for (int i = 0; i < 8; ++i) { t1 += a1[i]; t2 += a2[i]; }
  const float mu  = t1 * (1.0f / 512.0f);
  const float var = t2 * (1.0f / 512.0f) - mu * mu;
  const float rs  = rsqrtf(var + 1e-5f);
  const float o0 = (v0 - mu) * rs * sc[tid]       + bi[tid];
  const float o1 = (v1 - mu) * rs * sc[tid + 256] + bi[tid + 256];
  outH[(size_t)s * D_ + tid]       = (_Float16)o0;
  outH[(size_t)s * D_ + tid + 256] = (_Float16)o1;
  if constexpr (WF32) {
    outF[(size_t)s * D_ + tid]       = o0;
    outF[(size_t)s * D_ + tid + 256] = o1;
  }
}

// kv[b,h] = sum_s k(s) (x) v(s) stored transposed f16 [m][d]; z = sum_s k(s)
__global__ __launch_bounds__(256)
void k_kv_reduce(const _Float16* __restrict__ kh, const _Float16* __restrict__ vh,
                 _Float16* __restrict__ kvt, float* __restrict__ z)
{
  const int bh = blockIdx.x;            // 0..31
  const int b = bh >> 3, hh = bh & 7;
  const int tid = threadIdx.x;
  __shared__ float sk[8][64], sv[8][64];
  float acc[16];
#pragma unroll
  for (int j = 0; j < 16; ++j) acc[j] = 0.0f;
  float zacc = 0.0f;
  const int m = tid >> 2, db = (tid & 3) << 4;   // thread owns kv[m][db..db+15]
  const size_t base = (size_t)b * SEQ * D_ + hh * DH;
  for (int s0 = 0; s0 < SEQ; s0 += 8) {
    const int rr = tid >> 5, cc = (tid & 31) * 2;
    const size_t ro = base + (size_t)(s0 + rr) * D_;
    sk[rr][cc] = (float)kh[ro + cc]; sk[rr][cc + 1] = (float)kh[ro + cc + 1];
    sv[rr][cc] = (float)vh[ro + cc]; sv[rr][cc + 1] = (float)vh[ro + cc + 1];
    __syncthreads();
#pragma unroll
    for (int s = 0; s < 8; ++s) {
      const float vm = sv[s][m];
#pragma unroll
      for (int j = 0; j < 16; ++j) acc[j] += sk[s][db + j] * vm;
      if (tid < 64) zacc += sk[s][tid];
    }
    __syncthreads();
  }
  _Float16* o = kvt + (size_t)bh * 4096 + m * 64 + db;
#pragma unroll
  for (int j = 0; j < 16; ++j) o[j] = (_Float16)acc[j];
  if (tid < 64) z[bh * 64 + tid] = zacc;
}

// den = q . z per (token, head); attn = num / (den + 1e-6) as f16
__global__ __launch_bounds__(256)
void k_attn_div(const float* __restrict__ num, const _Float16* __restrict__ qh,
                const float* __restrict__ z, _Float16* __restrict__ attn)
{
  const int s = blockIdx.x;
  const int b = s >> 11;
  const int lane = threadIdx.x & 31, w = threadIdx.x >> 5;   // head = w
  const size_t o0 = (size_t)s * D_ + w * DH + lane;
  const float* zz = z + (b * 8 + w) * 64;
  const float q0 = (float)qh[o0], q1 = (float)qh[o0 + 32];
  float den = q0 * zz[lane] + q1 * zz[lane + 32];
#pragma unroll
  for (int off = 16; off; off >>= 1) den += __shfl_xor(den, off, 32);
  const float inv = 1.0f / (den + 1e-6f);
  attn[o0]      = (_Float16)(num[o0] * inv);
  attn[o0 + 32] = (_Float16)(num[o0 + 32] * inv);
}

// y[t, 0..7] = hn[t,:] @ proj_w + proj_b  (one wave per token)
__global__ __launch_bounds__(256)
void k_proj(const _Float16* __restrict__ hn, const float* __restrict__ pw,
            const float* __restrict__ pb, float* __restrict__ y)
{
  const int t = blockIdx.x * 8 + (threadIdx.x >> 5);
  const int lane = threadIdx.x & 31;
  float acc[8];
#pragma unroll
  for (int v = 0; v < 8; ++v) acc[v] = 0.0f;
  const _Float16* hr = hn + (size_t)t * D_;
  for (int d = lane; d < D_; d += 32) {
    const float hv = (float)hr[d];
#pragma unroll
    for (int v = 0; v < 8; ++v) acc[v] += hv * pw[d * 8 + v];
  }
#pragma unroll
  for (int v = 0; v < 8; ++v)
#pragma unroll
    for (int off = 16; off; off >>= 1) acc[v] += __shfl_xor(acc[v], off, 32);
  if (lane == 0) {
#pragma unroll
    for (int v = 0; v < 8; ++v) y[(size_t)t * 8 + v] = acc[v] + pb[v];
  }
}

// ---------------------------------------------------------------------------
extern "C" void kernel_launch(void* const* d_in, const int* in_sizes, int n_in,
                              void* d_out, int out_size, void* d_ws, size_t ws_size,
                              hipStream_t stream)
{
  (void)in_sizes; (void)n_in; (void)out_size; (void)ws_size;
  const int*   x     = (const int*)d_in[0];
  EmbTabs tabs;
  for (int i = 0; i < 8; ++i) tabs.t[i] = (const float*)d_in[1 + i];
  const float* in_w  = (const float*)d_in[9];
  const float* in_b  = (const float*)d_in[10];
  const float* ln1_s = (const float*)d_in[11];
  const float* ln1_b = (const float*)d_in[12];
  const float* wq    = (const float*)d_in[13];
  const float* wk    = (const float*)d_in[14];
  const float* wv    = (const float*)d_in[15];
  const float* wo    = (const float*)d_in[16];
  const float* bq    = (const float*)d_in[17];
  const float* bk    = (const float*)d_in[18];
  const float* bv    = (const float*)d_in[19];
  const float* bo    = (const float*)d_in[20];
  const float* ln2_s = (const float*)d_in[21];
  const float* ln2_b = (const float*)d_in[22];
  const float* w1    = (const float*)d_in[23];
  const float* b1    = (const float*)d_in[24];
  const float* w2    = (const float*)d_in[25];
  const float* b2    = (const float*)d_in[26];
  const float* lnf_s = (const float*)d_in[27];
  const float* lnf_b = (const float*)d_in[28];
  const float* pw    = (const float*)d_in[29];
  const float* pb    = (const float*)d_in[30];

  char* wp = (char*)d_ws;
  auto alloc = [&](size_t bytes) -> void* {
    void* p = (void*)wp;
    wp += (bytes + 255) & ~(size_t)255;
    return p;
  };
  _Float16* E    = (_Float16*)alloc((size_t)TOK * EMBW * 2);
  _Float16* WtIn = (_Float16*)alloc((size_t)D_ * EMBW * 2);
  _Float16* WtQ  = (_Float16*)alloc((size_t)LAYERS * D_ * D_ * 2);
  _Float16* WtK  = (_Float16*)alloc((size_t)LAYERS * D_ * D_ * 2);
  _Float16* WtV  = (_Float16*)alloc((size_t)LAYERS * D_ * D_ * 2);
  _Float16* WtO  = (_Float16*)alloc((size_t)LAYERS * D_ * D_ * 2);
  _Float16* Wt1  = (_Float16*)alloc((size_t)LAYERS * D_ * DFF_ * 2);
  _Float16* Wt2  = (_Float16*)alloc((size_t)LAYERS * D_ * DFF_ * 2);
  float*    h    = (float*)   alloc((size_t)TOK * D_ * 4);
  _Float16* xn   = (_Float16*)alloc((size_t)TOK * D_ * 2);
  _Float16* qh   = (_Float16*)alloc((size_t)TOK * D_ * 2);
  _Float16* kh   = (_Float16*)alloc((size_t)TOK * D_ * 2);
  _Float16* vh   = (_Float16*)alloc((size_t)TOK * D_ * 2);
  _Float16* kvt  = (_Float16*)alloc((size_t)32 * 4096 * 2);
  float*    z    = (float*)   alloc((size_t)32 * 64 * 4);
  float*    num  = (float*)   alloc((size_t)TOK * D_ * 4);
  _Float16* attn = (_Float16*)alloc((size_t)TOK * D_ * 2);
  _Float16* f1   = (_Float16*)alloc((size_t)TOK * DFF_ * 2);
  _Float16* hn   = (_Float16*)alloc((size_t)TOK * D_ * 2);

  auto cvt = [&](const float* s, _Float16* dT, int K, int N) {
    const int total = K * N;
    k_cvtT<<<dim3((total + 255) / 256), dim3(256), 0, stream>>>(s, dT, K, N);
  };

  // weights -> f16 transposed (inside the graph; ~230MB traffic ≈ 10us @23TB/s)
  cvt(in_w, WtIn, EMBW, D_);
  for (int l = 0; l < LAYERS; ++l) {
    cvt(wq + (size_t)l * D_ * D_,   WtQ + (size_t)l * D_ * D_,   D_,   D_);
    cvt(wk + (size_t)l * D_ * D_,   WtK + (size_t)l * D_ * D_,   D_,   D_);
    cvt(wv + (size_t)l * D_ * D_,   WtV + (size_t)l * D_ * D_,   D_,   D_);
    cvt(wo + (size_t)l * D_ * D_,   WtO + (size_t)l * D_ * D_,   D_,   D_);
    cvt(w1 + (size_t)l * D_ * DFF_, Wt1 + (size_t)l * D_ * DFF_, D_,   DFF_);
    cvt(w2 + (size_t)l * D_ * DFF_, Wt2 + (size_t)l * D_ * DFF_, DFF_, D_);
  }

  // embedding concat + input projection (+bias +posenc) -> h f32
  k_embed<<<dim3(TOK), dim3(256), 0, stream>>>(x, tabs, E);
  const dim3 gD(D_ / 128, TOK / 128);     // N=512 GEMMs
  const dim3 gF(DFF_ / 128, TOK / 128);   // N=2048 GEMM
  const dim3 gNum(1, SEQ / 128);          // per-(b,h) num GEMM, N=64
  k_gemm<EPI_POSENC, 128><<<gD, 256, 0, stream>>>(E, EMBW, WtIn, EMBW, in_b,
                                                  nullptr, h, nullptr, D_, EMBW);

  for (int l = 0; l < LAYERS; ++l) {
    const size_t wo2 = (size_t)l * D_ * D_;
    const size_t wf2 = (size_t)l * D_ * DFF_;
    k_ln<false><<<dim3(TOK), 256, 0, stream>>>(h, ln1_s + l * D_, ln1_b + l * D_,
                                               xn, nullptr);
    k_gemm<EPI_ELU1, 128><<<gD, 256, 0, stream>>>(xn, D_, WtQ + wo2, D_, bq + l * D_,
                                                  nullptr, nullptr, qh, D_, D_);
    k_gemm<EPI_ELU1, 128><<<gD, 256, 0, stream>>>(xn, D_, WtK + wo2, D_, bk + l * D_,
                                                  nullptr, nullptr, kh, D_, D_);
    k_gemm<EPI_F16, 128><<<gD, 256, 0, stream>>>(xn, D_, WtV + wo2, D_, bv + l * D_,
                                                 nullptr, nullptr, vh, D_, D_);
    k_kv_reduce<<<dim3(32), 256, 0, stream>>>(kh, vh, kvt, z);
    for (int bh = 0; bh < 32; ++bh) {
      const int b = bh >> 3, hd = bh & 7;
      const _Float16* qp = qh + (size_t)b * SEQ * D_ + hd * DH;
      float* np = num + (size_t)b * SEQ * D_ + hd * DH;
      k_gemm<EPI_F32, 64><<<gNum, 256, 0, stream>>>(qp, D_, kvt + (size_t)bh * 4096,
                                                    DH, nullptr, nullptr, np,
                                                    nullptr, D_, DH);
    }
    k_attn_div<<<dim3(TOK), 256, 0, stream>>>(num, qh, z, attn);
    k_gemm<EPI_RESCLIP, 128><<<gD, 256, 0, stream>>>(attn, D_, WtO + wo2, D_,
                                                     bo + l * D_, h, h, nullptr,
                                                     D_, D_);
    k_ln<false><<<dim3(TOK), 256, 0, stream>>>(h, ln2_s + l * D_, ln2_b + l * D_,
                                               xn, nullptr);
    k_gemm<EPI_GELU, 128><<<gF, 256, 0, stream>>>(xn, D_, Wt1 + wf2, D_,
                                                  b1 + l * DFF_, nullptr, nullptr,
                                                  f1, DFF_, D_);
    k_gemm<EPI_RESCLIP, 128><<<gD, 256, 0, stream>>>(f1, DFF_, Wt2 + wf2, DFF_,
                                                     b2 + l * D_, h, h, nullptr,
                                                     D_, DFF_);
  }

  // final LN: f32 -> d_out (h part) + f16 for the tiny projection
  k_ln<true><<<dim3(TOK), 256, 0, stream>>>(h, lnf_s, lnf_b, hn, (float*)d_out);
  k_proj<<<dim3(TOK / 8), 256, 0, stream>>>(hn, pw, pb,
                                            (float*)d_out + (size_t)TOK * D_);
}
